// Head_47614007443664
// MI455X (gfx1250) — compile-verified, matrix operand order checked by reading, not run on hardware
//
#include <hip/hip_runtime.h>

// Causal single-head attention for MI455X (gfx1250, wave32, WMMA).
//   B=4, T=4096, C=1024, HS=64.  fp32 in/out, f16 WMMA with f32 accumulate.
// Phase 1: fused K/Q/V projection (x read from HBM once), f16 K/Q row-major +
//          V transposed ([B][HS][T]) into d_ws.
// Phase 2: flash attention, one wave per 16-query tile, 32-key chunks,
//          double-buffered GLOBAL_LOAD_ASYNC_TO_LDS_B128 staging so the
//          ASYNCcnt DMA engine overlaps the WMMA pipeline.

#define TSEQ   4096
#define CDIM   1024
#define HSD    64
#define NBATCH 4

typedef _Float16 v16h __attribute__((ext_vector_type(16)));
typedef _Float16 v8h  __attribute__((ext_vector_type(8)));
typedef float    v8f  __attribute__((ext_vector_type(8)));
typedef int      v4i  __attribute__((ext_vector_type(4)));

#if defined(__has_builtin)
#if __has_builtin(__builtin_amdgcn_global_load_async_to_lds_b128) && \
    __has_builtin(__builtin_amdgcn_s_wait_asynccnt)
#define USE_ASYNC_LDS 1
#endif
#endif
#ifndef USE_ASYNC_LDS
#define USE_ASYNC_LDS 0
#endif

#if USE_ASYNC_LDS
// Builtin signature (probe-confirmed): (v4i AS1* src, v4i AS3* dst, imm, imm)
typedef __attribute__((address_space(1))) v4i g_v4i;
typedef __attribute__((address_space(3))) v4i l_v4i;
__device__ __forceinline__ void async_cp_b128(_Float16* dst_lds,
                                              const _Float16* src_glob) {
  __builtin_amdgcn_global_load_async_to_lds_b128(
      (g_v4i*)src_glob, (l_v4i*)dst_lds, 0, 0);
}
#endif

__device__ __forceinline__ v8f wmma16(v16h a, v16h b, v8f c) {
  // (neg_a, A, neg_b, B, c_mod, C, reuse_a, reuse_b)
  return __builtin_amdgcn_wmma_f32_16x16x32_f16(false, a, false, b, (short)0, c,
                                                false, false);
}

// Wave32 f16 fragment loader (2 x ds_load_b128 per fragment).
// A 16x32 row-major (ld in halves): A[m][k], lane m = lane&15,
//   k in {0..7,16..23} + 8*(lane>>4).
// B 32x16 stored TRANSPOSED in LDS (B[k][n] = base[n*ld + k]) uses the
//   identical per-lane pattern, so one helper serves both operands.
__device__ __forceinline__ v16h load_frag(const _Float16* base, int ld) {
  const int lane = threadIdx.x & 31;
  const _Float16* p = base + (lane & 15) * ld + ((lane >> 4) << 3);
  v8h lo = *(const v8h*)(p);
  v8h hi = *(const v8h*)(p + 16);
  v16h r;
#pragma unroll
  for (int i = 0; i < 8; ++i) { r[i] = lo[i]; r[i + 8] = hi[i]; }
  return r;
}

// ---------------------------------------------------------------------------
// Phase 1: K = x*Wk, Q = x*Wq*0.125, V = x*Wv   (f32 -> f16 outputs in d_ws)
// 256 blocks x 128 threads; block owns 64 rows of x; all three GEMMs fused so
// x (64 MB) is streamed from HBM exactly once.  V is stored transposed
// ([B][HS][T]) so phase 2 can stage it with straight b128 async copies.
// ---------------------------------------------------------------------------
__global__ __launch_bounds__(128) void proj_qkv(
    const float* __restrict__ x, const float* __restrict__ Wk,
    const float* __restrict__ Wq, const float* __restrict__ Wv,
    _Float16* __restrict__ Kh, _Float16* __restrict__ Qh,
    _Float16* __restrict__ Vt) {
  __shared__ __align__(16) _Float16 xs[64 * 64];      // 64 rows x 64 c (f16)
  __shared__ __align__(16) _Float16 wt[3][64 * 64];   // W chunks, h-major

  const int tid  = threadIdx.x;
  const int lane = tid & 31;
  const int wave = tid >> 5;
  const int rowbase = blockIdx.x * 64;

  const float* Ws[3] = {Wk, Wq, Wv};

  v8f acc[3][4] = {};

  for (int cb = 0; cb < CDIM; cb += 64) {
    // stage x tile (f32 -> f16), coalesced over columns
#pragma unroll 4
    for (int i = tid; i < 4096; i += 128) {
      int rr = i >> 6, cc = i & 63;
      xs[i] = (_Float16)x[(size_t)(rowbase + rr) * CDIM + cb + cc];
    }
    // stage W chunks transposed: wt[m][h*64 + c'] = W[cb+c'][h]
#pragma unroll
    for (int m = 0; m < 3; ++m) {
      const float* W = Ws[m];
#pragma unroll 4
      for (int i = tid; i < 4096; i += 128) {
        int cc = i >> 6, h = i & 63;
        wt[m][h * 64 + cc] = (_Float16)W[(size_t)(cb + cc) * HSD + h];
      }
    }
    __syncthreads();

    const _Float16* arow = xs + wave * 16 * 64;
#pragma unroll
    for (int kc = 0; kc < 2; ++kc) {
      v16h a = load_frag(arow + kc * 32, 64);
      // 12-WMMA chain with a double-buffered B fragment so several
      // ds_load_b128s stay in flight per wait (s_wait_dscnt > 0).
      v16h bcur = load_frag(&wt[0][0 * 16 * 64 + kc * 32], 64);
#pragma unroll
      for (int idx = 0; idx < 12; ++idx) {
        const int m = idx >> 2, t = idx & 3;
        v16h bnext = bcur;
        if (idx < 11) {
          const int m2 = (idx + 1) >> 2, t2 = (idx + 1) & 3;
          bnext = load_frag(&wt[m2][t2 * 16 * 64 + kc * 32], 64);
        }
        acc[m][t] = wmma16(a, bcur, acc[m][t]);
        bcur = bnext;
      }
    }
    __syncthreads();
  }

  // C/D layout: col = lane&15, rows rowoff..rowoff+7 in VGPRs 0..7
  const int col = lane & 15, rowoff = (lane >> 4) << 3;
  const int rw = rowbase + wave * 16 + rowoff;   // global row of acc[*][*][0]
  const int bb = rw >> 12;                       // batch (blocks never straddle)
  const int sq = rw & (TSEQ - 1);                // row within sequence
#pragma unroll
  for (int t = 0; t < 4; ++t)
#pragma unroll
    for (int r = 0; r < 8; ++r) {
      Kh[(size_t)(rw + r) * HSD + t * 16 + col] = (_Float16)acc[0][t][r];
      Qh[(size_t)(rw + r) * HSD + t * 16 + col] =
          (_Float16)(acc[1][t][r] * 0.125f);     // fold hs^-0.5 into Q
      // V transposed: Vt[b][h][seq]
      Vt[((size_t)bb * HSD + t * 16 + col) * TSEQ + sq + r] =
          (_Float16)acc[2][t][r];
    }
}

// Stage one 32-key K/V chunk (16 b128 transfers per wave).
__device__ __forceinline__ void stage_chunk(_Float16* ks, _Float16* vts,
                                            const _Float16* ksrc,
                                            const _Float16* vsrc, int lane) {
#if USE_ASYNC_LDS
#pragma unroll
  for (int i = lane; i < 256; i += 32)
    async_cp_b128(ks + i * 8, ksrc + i * 8);
#pragma unroll
  for (int i = lane; i < 256; i += 32) {
    int h = i >> 2, part = i & 3;
    async_cp_b128(vts + h * 32 + part * 8, vsrc + (size_t)h * TSEQ + part * 8);
  }
#else
#pragma unroll
  for (int i = lane; i < 256; i += 32)
    *(v8h*)(ks + i * 8) = *(const v8h*)(ksrc + i * 8);
#pragma unroll
  for (int i = lane; i < 256; i += 32) {
    int h = i >> 2, part = i & 3;
    *(v8h*)(vts + h * 32 + part * 8) =
        *(const v8h*)(vsrc + (size_t)h * TSEQ + part * 8);
  }
#endif
}

// ---------------------------------------------------------------------------
// Phase 2: flash attention. One wave per 16-query tile (1024 tiles total),
// 2 independent waves per block (private LDS slices -> no barriers; same-wave
// DS ops are in order per the CDNA5 ISA). Keys processed in chunks of 32 with
// ping-pong K/V buffers: chunk j+1's async DMA is issued before computing
// chunk j, and `s_wait_asynccnt 16` (async loads complete in order) ensures
// only the previous chunk has landed while the next streams in.
// ---------------------------------------------------------------------------
__global__ __launch_bounds__(64) void attn_flash(
    const _Float16* __restrict__ Qh, const _Float16* __restrict__ Kh,
    const _Float16* __restrict__ Vt, float* __restrict__ out) {
  // per wave: q 1024 | k0 2048 | k1 2048 | v0 2048 | v1 2048 | p 512 = 9728
  __shared__ __align__(16) _Float16 lds[2][9728];
  const int lane = threadIdx.x & 31;
  const int wave = threadIdx.x >> 5;
  _Float16* qs  = lds[wave];
  _Float16* kb0 = qs + 1024;
  _Float16* kb1 = kb0 + 2048;
  _Float16* vb0 = kb1 + 2048;
  _Float16* vb1 = vb0 + 2048;
  _Float16* ps  = vb1 + 2048;

  const int tile  = blockIdx.x * 2 + wave;
  const int b     = tile >> 8;          // 256 tiles per batch
  const int qbase = (tile & 255) << 4;
  const int qEnd  = qbase + 15;

  const _Float16* Qb = Qh + ((size_t)b * TSEQ + qbase) * HSD;
  const _Float16* Kb = Kh + (size_t)b * TSEQ * HSD;
  const _Float16* Vb = Vt + (size_t)b * HSD * TSEQ;

  // prologue: stage Q (4 DMAs) then chunk 0 (16 DMAs); waiting asynccnt<=16
  // drains the Q transfers while chunk 0 is still in flight.
#if USE_ASYNC_LDS
#pragma unroll
  for (int i = lane; i < 128; i += 32) async_cp_b128(qs + i * 8, Qb + i * 8);
  stage_chunk(kb0, vb0, Kb, Vb, lane);
  __builtin_amdgcn_s_wait_asynccnt(16);
#else
#pragma unroll
  for (int i = lane; i < 128; i += 32)
    *(v8h*)(qs + i * 8) = *(const v8h*)(Qb + i * 8);
  stage_chunk(kb0, vb0, Kb, Vb, lane);
#endif
  v16h aQ0 = load_frag(qs, 64);        // h 0..31
  v16h aQ1 = load_frag(qs + 32, 64);   // h 32..63

  v8f o[4] = {};
  float mrow[8], lrow[8];
#pragma unroll
  for (int r = 0; r < 8; ++r) { mrow[r] = -1e30f; lrow[r] = 0.0f; }

  const int col = lane & 15, rowoff = (lane >> 4) << 3;

  for (int jj = 0; jj <= qEnd; jj += 32) {
    const int cur = (jj >> 5) & 1;
    _Float16* ks  = cur ? kb1 : kb0;
    _Float16* vts = cur ? vb1 : vb0;

    // kick off next chunk into the other buffer, then wait for this chunk
    if (jj + 32 <= qEnd) {
      stage_chunk(cur ? kb0 : kb1, cur ? vb0 : vb1,
                  Kb + (size_t)(jj + 32) * HSD, Vb + jj + 32, lane);
#if USE_ASYNC_LDS
      __builtin_amdgcn_s_wait_asynccnt(16);  // allow next chunk in flight
#endif
    } else {
#if USE_ASYNC_LDS
      __builtin_amdgcn_s_wait_asynccnt(0);
#endif
    }

    // S = Q . K^T  (two 16x16 key tiles, K-dim 64 = 2 wmma accumulations each)
    v8f s0 = {}, s1 = {};
#pragma unroll
    for (int hc = 0; hc < 2; ++hc) {
      v16h a   = hc ? aQ1 : aQ0;
      v16h bk0 = load_frag(ks + hc * 32, 64);            // keys jj..jj+15
      v16h bk1 = load_frag(ks + 16 * 64 + hc * 32, 64);  // keys jj+16..jj+31
      s0 = wmma16(a, bk0, s0);
      s1 = wmma16(a, bk1, s1);
    }

    // causal mask (only the last chunk(s) can straddle the diagonal)
    if (jj + 31 > qbase) {
#pragma unroll
      for (int r = 0; r < 8; ++r) {
        int qrow = qbase + rowoff + r;
        if (jj + col > qrow)      s0[r] = -1e30f;
        if (jj + 16 + col > qrow) s1[r] = -1e30f;
      }
    }

    // online softmax: row reductions across 16 lanes (shfl_xor stays in half)
#pragma unroll
    for (int r = 0; r < 8; ++r) {
      float mx = fmaxf(s0[r], s1[r]);
      mx = fmaxf(mx, __shfl_xor(mx, 1));
      mx = fmaxf(mx, __shfl_xor(mx, 2));
      mx = fmaxf(mx, __shfl_xor(mx, 4));
      mx = fmaxf(mx, __shfl_xor(mx, 8));
      float mnew = fmaxf(mrow[r], mx);
      float a_   = __expf(mrow[r] - mnew);
      mrow[r] = mnew;
      float p0 = __expf(s0[r] - mnew);
      float p1 = __expf(s1[r] - mnew);
      float rs = p0 + p1;
      rs += __shfl_xor(rs, 1);
      rs += __shfl_xor(rs, 2);
      rs += __shfl_xor(rs, 4);
      rs += __shfl_xor(rs, 8);
      lrow[r] = lrow[r] * a_ + rs;
      // C-layout -> LDS so we can reload as an A fragment
      ps[(rowoff + r) * 32 + col]      = (_Float16)p0;
      ps[(rowoff + r) * 32 + 16 + col] = (_Float16)p1;
#pragma unroll
      for (int t = 0; t < 4; ++t) o[t][r] *= a_;
    }

    // O += P . V   (A = P 16x32, B = V via h-major LDS tile, 4 h-tiles)
    v16h aP = load_frag(ps, 32);
    v16h bvcur = load_frag(vts, 32);
#pragma unroll
    for (int t = 0; t < 4; ++t) {
      v16h bvnext = bvcur;
      if (t < 3) bvnext = load_frag(vts + (t + 1) * 16 * 32, 32);
      o[t] = wmma16(aP, bvcur, o[t]);
      bvcur = bvnext;
    }
  }

  // epilogue: normalize and store fp32 output
#pragma unroll
  for (int r = 0; r < 8; ++r) {
    float inv = 1.0f / lrow[r];
    size_t row = (size_t)b * TSEQ + qbase + rowoff + r;
#pragma unroll
    for (int t = 0; t < 4; ++t)
      out[row * HSD + t * 16 + col] = o[t][r] * inv;
  }
}

extern "C" void kernel_launch(void* const* d_in, const int* in_sizes, int n_in,
                              void* d_out, int out_size, void* d_ws,
                              size_t ws_size, hipStream_t stream) {
  const float* x  = (const float*)d_in[0];
  const float* Wk = (const float*)d_in[1];
  const float* Wq = (const float*)d_in[2];
  const float* Wv = (const float*)d_in[3];

  // workspace: K,Q row-major + V transposed, f16, 2 MB each (L2-resident)
  _Float16* Kh = (_Float16*)d_ws;
  _Float16* Qh = Kh + (size_t)NBATCH * TSEQ * HSD;
  _Float16* Vt = Qh + (size_t)NBATCH * TSEQ * HSD;

  proj_qkv<<<dim3((NBATCH * TSEQ) / 64), dim3(128), 0, stream>>>(
      x, Wk, Wq, Wv, Kh, Qh, Vt);
  attn_flash<<<dim3((NBATCH * TSEQ / 16) / 2), dim3(64), 0, stream>>>(
      Qh, Kh, Vt, (float*)d_out);
}
